// GCN_51015621542485
// MI455X (gfx1250) — matrix operand order repaired and checked by reading, not compiled, not run on hardware
//
#include <hip/hip_runtime.h>
#include <hip/hip_bf16.h>

// ---------------------------------------------------------------------------
// GCN 2-layer forward for MI455X (gfx1250, wave32, WMMA).
//   layer: h = X @ W  (bf16 WMMA, f32 accum)  ->  agg = b + dinv^2*h (+self loop)
//          scatter-add over edges with symmetric norm (f32 atomics, L2-resident)
//
// GEMM blocking: 160 rows x OUT_C per block (100000/160 = 625 blocks, exact).
// Per wave: B fragments held in registers (4 x v16bf), swept over row tiles
// -> 40 (layer1) / 20 (layer2) v_wmma per wave, 1 x 32B ds_load per WMMA.
// ---------------------------------------------------------------------------

typedef __attribute__((ext_vector_type(16))) __bf16 v16bf;
typedef __attribute__((ext_vector_type(8)))  float  v8f;

#define N_NODES 100000
#define IN_C    128
#define MTILE   160              // rows per block; 100000 % 160 == 0
#define RTILES  (MTILE / 16)     // 10 row tiles per block

// ---------------------------------------------------------------------------
// GEMM: H[MTILE x OUT_C] tile per block = X[MTILE x 128] @ W[128 x OUT_C]
// A staged in LDS pre-swizzled into per-lane A fragments, contiguous k = 16*(lane/16) + elem;
// W staged transposed (column-major bf16) so B fragments are contiguous 32B.
// blockDim.x == 256 (8 waves). For OUT_C=128: wave = column tile, 10 row
// tiles each. For OUT_C=64: wave&3 = column tile, wave>>2 = row half (5 each).
// ---------------------------------------------------------------------------
template <int OUT_C, bool RELU>
__global__ __launch_bounds__(256) void gcn_gemm_wmma(
    const float* __restrict__ X, const float* __restrict__ W,
    float* __restrict__ H)
{
    constexpr int COL_TILES = OUT_C / 16;       // 8 or 4
    constexpr int ROW_SPLIT = 8 / COL_TILES;    // 1 or 2
    constexpr int RT_PER    = RTILES / ROW_SPLIT; // 10 or 5

    __shared__ alignas(32) __bf16 As[RTILES][4][32][16]; // [row-tile][k-step][lane][elem]
    __shared__ alignas(32) __bf16 Bs[OUT_C][IN_C];       // W transposed, bf16

    const int tid  = threadIdx.x;
    const int row0 = blockIdx.x * MTILE;

    // ---- stage A (MTILE x 128 fp32 -> bf16, fragment-swizzled) ----
    for (int t = tid; t < MTILE * IN_C; t += 256) {
        const int m = t >> 7;           // row in block tile (0..159)
        const int k = t & 127;          // K index
        float v = X[(size_t)(row0 + m) * IN_C + k];
        if (RELU) v = fmaxf(v, 0.0f);
        const int mt = m >> 4;          // row tile
        const int ml = m & 15;          // row within tile
        const int kk = k >> 5;          // which 16x16x32 step
        const int kr = k & 31;          // K within step
        const int ln = ml + ((kr >> 4) << 4);            // lane
        const int el = kr & 15;                          // elem 0..15
        As[mt][kk][ln][el] = (__bf16)v;
    }
    // ---- stage B: Bs[n][k] = W[k][n] as bf16 ----
    for (int t = tid; t < IN_C * OUT_C; t += 256) {
        const int k = t / OUT_C;
        const int n = t % OUT_C;
        Bs[n][k] = (__bf16)W[(size_t)k * OUT_C + n];
    }
    __syncthreads();

    const int wave = tid >> 5;
    const int lane = tid & 31;
    const int ct   = wave % COL_TILES;          // column tile
    const int rt0  = (wave / COL_TILES) * RT_PER; // first row tile

    const int ncol  = ct * 16 + (lane & 15);    // output column (B layout)
    const int kbase = (lane >> 4) * 16;         // B lane-group K offset

    // B fragments live in registers across all row tiles
    v16bf bfrag[4];
#pragma unroll
    for (int kk = 0; kk < 4; ++kk)
        bfrag[kk] = *(const v16bf*)(&Bs[ncol][kk * 32 + kbase]);

    const int rbase = (lane >> 4) << 3;         // C/D row offset for this lane
    float* dcol = H + (size_t)row0 * OUT_C + ct * 16 + (lane & 15);

#pragma unroll
    for (int mt = rt0; mt < rt0 + RT_PER; ++mt) {
        v8f acc = {};
#pragma unroll
        for (int kk = 0; kk < 4; ++kk) {
            v16bf a = *(const v16bf*)(&As[mt][kk][lane][0]);
            acc = __builtin_amdgcn_wmma_f32_16x16x32_bf16(
                false, a, false, bfrag[kk], (short)0, acc, false, false);
        }
        float* dst = dcol + (size_t)(mt * 16 + rbase) * OUT_C;
#pragma unroll
        for (int r = 0; r < 8; ++r)
            dst[(size_t)r * OUT_C] = acc[r];
    }
}

// ---------------------------------------------------------------------------
// Normalization helpers
// ---------------------------------------------------------------------------
__global__ void k_fill_deg(float* __restrict__ deg, int n) {
    int i = blockIdx.x * blockDim.x + threadIdx.x;
    if (i < n) deg[i] = 1.0f;                         // self-loop
}
__global__ void k_count_deg(const int* __restrict__ dst, float* __restrict__ deg, int e) {
    int i = blockIdx.x * blockDim.x + threadIdx.x;
    if (i < e) atomicAdd(&deg[dst[i]], 1.0f);
}
__global__ void k_rsqrt_inplace(float* __restrict__ deg, int n) {
    int i = blockIdx.x * blockDim.x + threadIdx.x;
    if (i < n) {
        float d = deg[i];
        deg[i] = (d > 0.0f) ? rsqrtf(d) : 0.0f;       // deg -> dinv
    }
}

// agg[i,c] = bias[c] + h[i,c] * dinv[i]^2   (bias + self-loop message)
template <int CH>
__global__ void k_init_agg(const float* __restrict__ h, const float* __restrict__ dinv,
                           const float* __restrict__ bias, float* __restrict__ agg, int n)
{
    int idx = blockIdx.x * blockDim.x + threadIdx.x;
    if (idx >= n * CH) return;
    const int node = idx / CH;                        // CH is pow2 -> shifts
    const int c    = idx % CH;
    const float di = dinv[node];
    agg[idx] = bias[c] + h[idx] * di * di;
}

// one wave per edge; lane owns CH/32 contiguous channels (coalesced row burst)
template <int CH>
__global__ void k_scatter(const float* __restrict__ h, const int* __restrict__ src,
                          const int* __restrict__ dst, const float* __restrict__ dinv,
                          float* __restrict__ agg, int e)
{
    const int gid  = blockIdx.x * blockDim.x + threadIdx.x;
    const int edge = gid >> 5;
    const int lane = gid & 31;
    if (edge >= e) return;
    const int s = src[edge];
    const int d = dst[edge];
    // kick the gathered row toward the WGP while the dependent dinv loads run
    __builtin_prefetch(h + (size_t)s * CH + lane * (CH / 32), 0, 0);
    const float norm = dinv[s] * dinv[d];
    constexpr int PER = CH / 32;                      // 4 (CH=128) or 2 (CH=64)
    const float* hp = h   + (size_t)s * CH + lane * PER;
    float*       ap = agg + (size_t)d * CH + lane * PER;
#pragma unroll
    for (int j = 0; j < PER; ++j)
        atomicAdd(ap + j, hp[j] * norm);              // f32 atomic at L2
}

// ---------------------------------------------------------------------------
extern "C" void kernel_launch(void* const* d_in, const int* in_sizes, int n_in,
                              void* d_out, int out_size, void* d_ws, size_t ws_size,
                              hipStream_t stream)
{
    (void)n_in; (void)out_size; (void)ws_size;

    const float* x   = (const float*)d_in[0];
    const int*   ei  = (const int*)d_in[1];
    const float* W1  = (const float*)d_in[2];
    const float* b1  = (const float*)d_in[3];
    const float* W2  = (const float*)d_in[4];
    const float* b2  = (const float*)d_in[5];
    float*       out = (float*)d_out;

    const int N = N_NODES;
    const int E = in_sizes[1] / 2;
    const int* esrc = ei;
    const int* edst = ei + E;

    // carve workspace
    char*  base = (char*)d_ws;
    size_t off  = 0;
    auto carve = [&](size_t bytes) -> float* {
        float* p = (float*)(base + off);
        off += (bytes + 255) & ~(size_t)255;
        return p;
    };
    float* dinv = carve((size_t)N * sizeof(float));            // deg -> dinv
    float* h1   = carve((size_t)N * 128 * sizeof(float));
    float* agg1 = carve((size_t)N * 128 * sizeof(float));
    float* h2   = carve((size_t)N * 64  * sizeof(float));

    const int T = 256;
    // ---- symmetric normalization ----
    k_fill_deg     <<<(N + T - 1) / T, T, 0, stream>>>(dinv, N);
    k_count_deg    <<<(E + T - 1) / T, T, 0, stream>>>(edst, dinv, E);
    k_rsqrt_inplace<<<(N + T - 1) / T, T, 0, stream>>>(dinv, N);

    // ---- layer 1: h1 = x @ W1 ; agg1 = b1 + dinv^2 h1 + scatter ----
    gcn_gemm_wmma<128, false><<<N / MTILE, 256, 0, stream>>>(x, W1, h1);
    k_init_agg<128><<<((size_t)N * 128 + T - 1) / T, T, 0, stream>>>(h1, dinv, b1, agg1, N);
    k_scatter <128><<<((size_t)E * 32 + T - 1) / T, T, 0, stream>>>(h1, esrc, edst, dinv, agg1, E);

    // ---- layer 2: h2 = relu(agg1) @ W2 ; out = b2 + dinv^2 h2 + scatter ----
    gcn_gemm_wmma<64, true><<<N / MTILE, 256, 0, stream>>>(agg1, W2, h2);
    k_init_agg<64><<<((size_t)N * 64 + T - 1) / T, T, 0, stream>>>(h2, dinv, b2, out, N);
    k_scatter <64><<<((size_t)E * 32 + T - 1) / T, T, 0, stream>>>(h2, esrc, edst, dinv, out, E);
}
